// CDs2Hg_22419729285957
// MI455X (gfx1250) — matrix operand order repaired and checked by reading, not compiled, not run on hardware
//
#include <hip/hip_runtime.h>
#include <hip/hip_bf16.h>

// ---------------------------------------------------------------------------
// Problem constants (from reference)
// ---------------------------------------------------------------------------
#define N_ROWS   16384
#define N_CLS    1024
#define N_FEAT   256
#define K_VIRT   512            // [Xn^2 | Xn] concatenated
#define TAU_C    32.0f
#define ALPHA_C  0.9f

typedef __attribute__((ext_vector_type(16))) __bf16 v16bf;
typedef __attribute__((ext_vector_type(8)))  float  v8f;

__device__ __forceinline__ v8f wmma_bf16(v16bf a, v16bf b, v8f c) {
    // D = A(16x32 bf16) x B(32x16 bf16) + C(16x16 f32)
    return __builtin_amdgcn_wmma_f32_16x16x32_bf16(
        /*neg_a=*/false, a, /*neg_b=*/false, b,
        /*c_mod=*/(short)0, c, /*reuse_a=*/false, /*reuse_b=*/false);
}

__device__ __forceinline__ unsigned short bf_bits(__bf16 b) {
    union { __bf16 b; unsigned short u; } cv; cv.b = b; return cv.u;
}

// ---------------------------------------------------------------------------
// Kernel 0a: zero class mask + loss accumulators
// ---------------------------------------------------------------------------
__global__ void zero_kernel(float* __restrict__ maskv, float* __restrict__ lossAcc) {
    int i = blockIdx.x * blockDim.x + threadIdx.x;
    if (i < N_CLS) maskv[i] = 0.0f;
    if (i < 2)     lossAcc[i] = 0.0f;
}

// Kernel 0b: mask[T[n]] = 1 (racy same-value stores are benign)
__global__ void scatter_kernel(const int* __restrict__ T, float* __restrict__ maskv) {
    int n = blockIdx.x * blockDim.x + threadIdx.x;
    if (n < N_ROWS) maskv[T[n]] = 1.0f;
}

// Kernel 0c: per-row 1/||X[n]|| (one block per row, 256 threads = F)
__global__ __launch_bounds__(256) void xnorm_kernel(const float* __restrict__ X,
                                                    float* __restrict__ invX) {
    __shared__ float red[256];
    int n = blockIdx.x, f = threadIdx.x;
    float x = X[n * N_FEAT + f];
    red[f] = x * x;
    __syncthreads();
    for (int s = 128; s > 0; s >>= 1) {
        if (f < s) red[f] += red[f + s];
        __syncthreads();
    }
    if (f == 0) invX[n] = 1.0f / fmaxf(sqrtf(red[0]), 1e-12f);
}

// ---------------------------------------------------------------------------
// Kernel 1: per-class precompute.
//   B[c, k]       = inv_cov[c,k]               (k <  256)
//   B[c, 256+k]   = -2 * mun[c,k]*inv_cov[c,k] (k >= 256)
//   split each f32 into bf16 hi + bf16 lo ; Kc[c] = sum mun^2*inv_cov
// ---------------------------------------------------------------------------
__global__ __launch_bounds__(256) void classes_kernel(const float* __restrict__ means,
                                                      const float* __restrict__ logv,
                                                      unsigned short* __restrict__ Bhi,
                                                      unsigned short* __restrict__ Blo,
                                                      float* __restrict__ Kc) {
    __shared__ float red[256];
    int c = blockIdx.x, f = threadIdx.x;
    float mu = means[c * N_FEAT + f];
    red[f] = mu * mu;
    __syncthreads();
    for (int s = 128; s > 0; s >>= 1) {
        if (f < s) red[f] += red[f + s];
        __syncthreads();
    }
    float inv = 1.0f / fmaxf(sqrtf(red[0]), 1e-12f);
    __syncthreads();

    float lv  = fminf(fmaxf(logv[c * N_FEAT + f], 0.0f), 6.0f);   // relu6
    float ic  = __expf(-lv);
    float mun = mu * inv;
    float b0  = ic;
    float b1  = -2.0f * mun * ic;

    __bf16 h0 = (__bf16)b0; __bf16 l0 = (__bf16)(b0 - (float)h0);
    __bf16 h1 = (__bf16)b1; __bf16 l1 = (__bf16)(b1 - (float)h1);
    Bhi[c * K_VIRT + f]          = bf_bits(h0);
    Blo[c * K_VIRT + f]          = bf_bits(l0);
    Bhi[c * K_VIRT + N_FEAT + f] = bf_bits(h1);
    Blo[c * K_VIRT + N_FEAT + f] = bf_bits(l1);

    red[f] = mun * mun * ic;
    __syncthreads();
    for (int s = 128; s > 0; s >>= 1) {
        if (f < s) red[f] += red[f + s];
        __syncthreads();
    }
    if (f == 0) Kc[c] = red[0];
}

// ---------------------------------------------------------------------------
// Kernel 2: fused WMMA GEMM (dist) + softmax loss + H epilogue.
//   grid = N/16 workgroups; block = 256 threads = 8 wave32.
//   WG computes 16 rows x 1024 cols; wave w owns column tiles w*8..w*8+7.
// ---------------------------------------------------------------------------
#define WAVES 8
#define TPW   8   // 16x16 tiles per wave along columns

__global__ __launch_bounds__(256) void gmm_main(const float* __restrict__ X,
                                                const int* __restrict__ T,
                                                const float* __restrict__ invX,
                                                const unsigned short* __restrict__ Bhi,
                                                const unsigned short* __restrict__ Blo,
                                                const float* __restrict__ Kc,
                                                const float* __restrict__ maskv,
                                                float* __restrict__ lossAcc,
                                                float* __restrict__ out) {
    __shared__ int   Tl[16];
    __shared__ float maxbuf[16][WAVES];
    __shared__ float sumbuf[16][WAVES];
    __shared__ float stl[16];
    __shared__ float rowmaxl[16];

    const int tid  = threadIdx.x;
    const int w    = tid >> 5;        // wave id (wave32)
    const int lane = tid & 31;
    const int ln   = lane & 15;       // low lane: M row (A) / N col (B/D)
    const int h    = lane >> 4;       // lane half
    const int rb   = blockIdx.x * 16; // row base

    if (tid < 16) Tl[tid] = T[rb + tid];
    __syncthreads();

    const int   rowA = rb + ln;       // A-matrix row held by this lane
    const float inv  = invX[rowA];

    v8f acc[TPW] = {};                // 8 x (16x16 f32) accumulators

    union FragU { v16bf v; uint4 q[2]; };

    // Per-lane base pointers (kk advances by immediate offsets under unroll)
    const float*          xrow  = X + (size_t)rowA * N_FEAT + h * 16;
    const unsigned short* bhrow = Bhi + (size_t)ln * K_VIRT + h * 16; // + cB tile offset
    const unsigned short* blrow = Blo + (size_t)ln * K_VIRT + h * 16;

    // One K-step of 32 over one half of the virtual K (SQ: use Xn^2, else Xn).
    // PF: statically-known prefetch step (one 128B line = next two K-steps).
    auto kstep = [&](int kk, bool SQ, bool PF) {
        const int fbase = (SQ ? kk : kk - N_FEAT);

        // A fragment (16x32 bf16): lane holds K chunk [c0..c0+15] (c0 = 16*h)
        const float* xp = xrow + fbase;
        float xv[16];
        *(float4*)&xv[0]  = ((const float4*)xp)[0];
        *(float4*)&xv[4]  = ((const float4*)xp)[1];
        *(float4*)&xv[8]  = ((const float4*)(xp + 8))[0];
        *(float4*)&xv[12] = ((const float4*)(xp + 8))[1];

        union { v16bf v; __bf16 e[16]; } Ah, Al;
        #pragma unroll
        for (int i = 0; i < 16; ++i) {
            float xn = xv[i] * inv;
            float a  = SQ ? xn * xn : xn;
            __bf16 hi = (__bf16)a;
            Ah.e[i] = hi;
            Al.e[i] = (__bf16)(a - (float)hi);
        }

        // B fragments, double-buffered so tile j+1 loads overlap tile j WMMAs.
        FragU Bh[2], Bl[2];
        auto loadB = [&](int j, FragU& bhF, FragU& blF) {
            const size_t off = (size_t)((w * TPW + j) * 16) * K_VIRT + kk;
            const unsigned short* bh = bhrow + off;
            const unsigned short* bl = blrow + off;
            if (PF) {
                // locality=3 -> WGP-scope prefetch (pull into all cache levels)
                __builtin_prefetch(bh + 64, 0, 3);
                __builtin_prefetch(bl + 64, 0, 3);
            }
            bhF.q[0] = ((const uint4*)bh)[0];
            bhF.q[1] = ((const uint4*)bh)[1];
            blF.q[0] = ((const uint4*)bl)[0];
            blF.q[1] = ((const uint4*)bl)[1];
        };

        loadB(0, Bh[0], Bl[0]);
        #pragma unroll
        for (int j = 0; j < TPW; ++j) {
            const int cur = j & 1, nxt = cur ^ 1;
            if (j + 1 < TPW) loadB(j + 1, Bh[nxt], Bl[nxt]);
            // split-precision accumulation: ah*bh + ah*bl + al*bh
            acc[j] = wmma_bf16(Ah.v, Bh[cur].v, acc[j]);
            acc[j] = wmma_bf16(Ah.v, Bl[cur].v, acc[j]);
            acc[j] = wmma_bf16(Al.v, Bh[cur].v, acc[j]);
        }
    };

    // K = 0..255  : A = Xn^2 vs B = inv_cov
    #pragma unroll 2
    for (int ks = 0; ks < 8; ++ks)
        kstep(ks * 32, /*SQ=*/true, /*PF=*/(ks & 1) == 0);
    // K = 256..511: A = Xn    vs B = -2*mun*inv_cov
    #pragma unroll 2
    for (int ks = 8; ks < 16; ++ks)
        kstep(ks * 32, /*SQ=*/false, /*PF=*/((ks & 1) == 0) && ks < 14);

    // Per-column constants
    float kc[TPW], mk[TPW];
    #pragma unroll
    for (int j = 0; j < TPW; ++j) {
        int col = (w * TPW + j) * 16 + ln;
        kc[j] = Kc[col];
        mk[j] = maskv[col];
    }

    // H epilogue (streaming, non-temporal) + capture s_t = -TAU*dist[row,T[row]]
    #pragma unroll
    for (int j = 0; j < TPW; ++j) {
        int col = (w * TPW + j) * 16 + ln;
        #pragma unroll
        for (int r = 0; r < 8; ++r) {
            int   rl  = r + 8 * h;                 // local row (D layout)
            float d   = acc[j][r] + kc[j];
            bool  isT = (col == Tl[rl]);
            if (isT) stl[rl] = -TAU_C * d;         // unique writer per row
            float hv = isT ? 1.0f : __expf(-ALPHA_C * d);
            __builtin_nontemporal_store(hv * mk[j],
                &out[1 + (size_t)(rb + rl) * N_CLS + col]);
        }
    }

    // Row max of s = -TAU*dist  (butterfly within 16-lane half, then LDS)
    float lm[8];
    #pragma unroll
    for (int r = 0; r < 8; ++r) {
        float m = -3.4e38f;
        #pragma unroll
        for (int j = 0; j < TPW; ++j) m = fmaxf(m, -TAU_C * (acc[j][r] + kc[j]));
        m = fmaxf(m, __shfl_xor(m, 1, 32));
        m = fmaxf(m, __shfl_xor(m, 2, 32));
        m = fmaxf(m, __shfl_xor(m, 4, 32));
        m = fmaxf(m, __shfl_xor(m, 8, 32));
        lm[r] = m;
    }
    if (ln == 0) {
        #pragma unroll
        for (int r = 0; r < 8; ++r) maxbuf[r + 8 * h][w] = lm[r];
    }
    __syncthreads();
    if (tid < 16) {
        float m = maxbuf[tid][0];
        for (int q = 1; q < WAVES; ++q) m = fmaxf(m, maxbuf[tid][q]);
        rowmaxl[tid] = m;
    }
    __syncthreads();

    // Row sum of exp(s - max)
    #pragma unroll
    for (int r = 0; r < 8; ++r) {
        float rm = rowmaxl[r + 8 * h];
        float s  = 0.0f;
        #pragma unroll
        for (int j = 0; j < TPW; ++j)
            s += __expf(-TAU_C * (acc[j][r] + kc[j]) - rm);
        s += __shfl_xor(s, 1, 32);
        s += __shfl_xor(s, 2, 32);
        s += __shfl_xor(s, 4, 32);
        s += __shfl_xor(s, 8, 32);
        if (ln == 0) sumbuf[r + 8 * h][w] = s;
    }
    __syncthreads();
    if (tid < 16) {
        float Z = 0.0f;
        for (int q = 0; q < WAVES; ++q) Z += sumbuf[tid][q];
        float pt = __expf(stl[tid] - rowmaxl[tid]) / Z;
        if (pt > 0.0f) {
            atomicAdd(&lossAcc[0], -__logf(pt));
            atomicAdd(&lossAcc[1], 1.0f);
        }
    }
}

// ---------------------------------------------------------------------------
// Kernel 3: finalize scalar loss
// ---------------------------------------------------------------------------
__global__ void finalize_kernel(const float* __restrict__ lossAcc, float* __restrict__ out) {
    if (threadIdx.x == 0 && blockIdx.x == 0)
        out[0] = lossAcc[0] / fmaxf(lossAcc[1], 1.0f);
}

// ---------------------------------------------------------------------------
// Host launcher
// ---------------------------------------------------------------------------
extern "C" void kernel_launch(void* const* d_in, const int* in_sizes, int n_in,
                              void* d_out, int out_size, void* d_ws, size_t ws_size,
                              hipStream_t stream) {
    const float* X     = (const float*)d_in[0];
    const int*   T     = (const int*)  d_in[1];
    const float* means = (const float*)d_in[2];
    const float* logv  = (const float*)d_in[3];
    float*       out   = (float*)d_out;

    // Workspace layout (~2.2 MB total)
    float*          invX    = (float*)d_ws;                    // [16384]
    float*          Kc      = invX + N_ROWS;                   // [1024]
    float*          maskv   = Kc + N_CLS;                      // [1024]
    float*          lossAcc = maskv + N_CLS;                   // [4] (2 used)
    unsigned short* Bhi     = (unsigned short*)(lossAcc + 4);  // [1024*512] bf16
    unsigned short* Blo     = Bhi + (size_t)N_CLS * K_VIRT;    // [1024*512] bf16

    zero_kernel    <<<dim3((N_CLS + 255) / 256), dim3(256), 0, stream>>>(maskv, lossAcc);
    scatter_kernel <<<dim3(N_ROWS / 256),        dim3(256), 0, stream>>>(T, maskv);
    xnorm_kernel   <<<dim3(N_ROWS),              dim3(256), 0, stream>>>(X, invX);
    classes_kernel <<<dim3(N_CLS),               dim3(256), 0, stream>>>(means, logv, Bhi, Blo, Kc);
    gmm_main       <<<dim3(N_ROWS / 16),         dim3(256), 0, stream>>>(X, T, invX, Bhi, Blo, Kc,
                                                                         maskv, lossAcc, out);
    finalize_kernel<<<dim3(1),                   dim3(1),   0, stream>>>(lossAcc, out);
}
